// Flow_PWC2_8134668058806
// MI455X (gfx1250) — matrix operand order repaired and checked by reading, not compiled, not run on hardware
//
#include <hip/hip_runtime.h>

// ---------------------------------------------------------------------------
// CDNA5 (gfx1250) PWC-Net style optical flow.
// 3x3 convs = implicit GEMM on v_wmma_f32_16x16x32_f16 (wave32).
//  - Activations mirrored to f16 NHWC (channel-contiguous, padded stride)
//  - GEMM K ordered tap-major with per-tap channel padding to 16, so each
//    half-wave's B fragment is ONE contiguous v16h load (2x global_load_b128)
//  - Weights pre-packed to f16 in exact A-fragment order -> one v16h load
//  - Out-of-range K (channel pad / tap pad) carries ZERO WEIGHTS in A, so
//    pad reads need no masking; spatial bounds are one per-chunk vector select
//  - One wave = 16 Cout x 32 pixels (two N-tiles reusing one A fragment):
//    1 A-load amortized over 2 WMMAs
// ---------------------------------------------------------------------------

typedef __attribute__((ext_vector_type(16))) _Float16 v16h;
typedef __attribute__((ext_vector_type(8)))  float    v8f;

#define LEAK 0.1f

__device__ __forceinline__ float lrelu_f(float v) { return v > 0.f ? v : LEAK * v; }

// A-fragment K index for element e, half-wave grp (contiguous: k = 16*grp + e, matches B)
__device__ __forceinline__ int a_klane(int e, int grp) {
    return grp * 16 + e;
}

// ---------------------------------------------------------------------------
// Weight pre-pack: f32 OIHW -> f16 fragment-ordered, tap-major padded K
//   k = tap * Cpad + ci ;  zero for ci >= Cin, tap >= 9, co >= Cout
// ---------------------------------------------------------------------------
__global__ __launch_bounds__(256) void packw_kernel(
    const float* __restrict__ wgt, _Float16* __restrict__ wp,
    int Cout, int Cin, int Cpad, int Mt, int Kc)
{
    const int t = blockIdx.x * blockDim.x + threadIdx.x;
    const int total = Mt * Kc * 32;
    if (t >= total) return;
    const int lane = t & 31;
    const int t2 = t >> 5;
    const int kc = t2 % Kc;
    const int mt = t2 / Kc;
    const int grp = lane >> 4, l16 = lane & 15;
    const int co = mt * 16 + l16;
    _Float16* dst = wp + (size_t)t * 16;
#pragma unroll
    for (int e = 0; e < 16; ++e) {
        const int k = kc * 32 + a_klane(e, grp);
        const int tap = k / Cpad;
        const int ci  = k % Cpad;
        const bool valid = (co < Cout) & (ci < Cin) & (tap < 9);
        const int coC = min(co, Cout - 1);
        const int ciC = min(ci, Cin - 1);
        const int tpC = min(tap, 8);
        const float x = wgt[((size_t)coC * Cin + ciC) * 9 + tpC];
        dst[e] = valid ? (_Float16)x : (_Float16)0.0f;
    }
}

// ---------------------------------------------------------------------------
// Implicit-GEMM 3x3 conv (stride/pad/dilation).
// in16: f16 NHWC mirror, per-pixel channel stride CinTotP, window at cinOff.
// out32: f32 NCHW (channel-strided CoutTot/coutOff), optional.
// out16: f16 NHWC mirror (stride CoutTotP), optional.
// One wave = 16 Cout x 32 pixels (two 16-pixel tiles share the A fragment).
// Bias + LeakyReLU fused into the writeback.
// ---------------------------------------------------------------------------
__global__ __launch_bounds__(256) void conv3x3_wmma(
    const _Float16* __restrict__ in16, int CinTotP, int cinOff,
    int Hin, int Win,
    const _Float16* __restrict__ wp, const float* __restrict__ bias,
    float* __restrict__ out32, int CoutTot,
    _Float16* __restrict__ out16, int CoutTotP,
    int coutOff, int Cout, int Hout, int Wout,
    int stride, int pad, int dil, int relu,
    int Mtiles, int Ntot, int Kc, int Cpad)
{
    const int lane = threadIdx.x & 31;
    const int wv   = threadIdx.x >> 5;
    const long long tile = (long long)blockIdx.x * 8 + wv;
    const int Ntiles2 = (Ntot + 31) >> 5;          // pairs of 16-pixel tiles
    const int mt = (int)(tile % Mtiles);
    const long long nt = tile / Mtiles;
    if (nt >= Ntiles2) return;

    const int grp = lane >> 4;
    const int l16 = lane & 15;
    const int hw = Hout * Wout;

    // decode both pixel tiles for this lane (B/D columns), clamped for safety
    bool nvA[2];
    int  iy0A[2], ix0A[2], rowA[2];
    long long pixOutA[2], out32A[2];
#pragma unroll
    for (int s = 0; s < 2; ++s) {
        const int n = (int)nt * 32 + s * 16 + l16;
        const bool nv = n < Ntot;
        const int nc = nv ? n : 0;
        const int bb  = nc / hw;
        const int rem = nc % hw;
        const int oy = rem / Wout, ox = rem % Wout;
        nvA[s]  = nv;
        iy0A[s] = oy * stride - pad;
        ix0A[s] = ox * stride - pad;
        rowA[s] = bb * Hin;
        pixOutA[s] = (long long)(bb * Hout + oy) * Wout + ox;
        out32A[s]  = (((long long)bb * CoutTot + coutOff) * Hout + oy) * Wout + ox;
    }

    const _Float16* ap = wp + ((size_t)mt * Kc * 32 + lane) * 16;

    // tap / channel-base within chunk k-window for this half-wave
    const int kg0 = grp * 16;
    int tap   = (kg0 >= Cpad) ? 1 : 0;
    int cbase = kg0 - (tap ? Cpad : 0);

    const v16h vzero = {};
    v8f acc0 = {}, acc1 = {};
    for (int kc = 0; kc < Kc; ++kc) {
        const v16h a = *(const v16h*)(ap + (size_t)kc * 512);

        const int ty = tap / 3, tx = tap - ty * 3;
        const bool tapok = tap < 9;

        // tile 0
        {
            const int iy = iy0A[0] + ty * dil;
            const int ix = ix0A[0] + tx * dil;
            const bool ok = nvA[0] & tapok &
                            ((unsigned)iy < (unsigned)Hin) &
                            ((unsigned)ix < (unsigned)Win);
            const int iyC = min(max(iy, 0), Hin - 1);
            const int ixC = min(max(ix, 0), Win - 1);
            const long long addr =
                ((long long)(rowA[0] + iyC) * Win + ixC) * CinTotP + cinOff + cbase;
            const v16h bv = *(const v16h*)(in16 + addr);
            const v16h bm = ok ? bv : vzero;
            acc0 = __builtin_amdgcn_wmma_f32_16x16x32_f16(
                false, a, false, bm, (short)0, acc0, false, false);
        }
        // tile 1
        {
            const int iy = iy0A[1] + ty * dil;
            const int ix = ix0A[1] + tx * dil;
            const bool ok = nvA[1] & tapok &
                            ((unsigned)iy < (unsigned)Hin) &
                            ((unsigned)ix < (unsigned)Win);
            const int iyC = min(max(iy, 0), Hin - 1);
            const int ixC = min(max(ix, 0), Win - 1);
            const long long addr =
                ((long long)(rowA[1] + iyC) * Win + ixC) * CinTotP + cinOff + cbase;
            const v16h bv = *(const v16h*)(in16 + addr);
            const v16h bm = ok ? bv : vzero;
            acc1 = __builtin_amdgcn_wmma_f32_16x16x32_f16(
                false, a, false, bm, (short)0, acc1, false, false);
        }

        cbase += 32;
        if (cbase >= Cpad) { cbase -= Cpad; ++tap; }
        if (cbase >= Cpad) { cbase -= Cpad; ++tap; }
    }

#pragma unroll
    for (int s = 0; s < 2; ++s) {
        if (!nvA[s]) continue;
        const v8f acc = s ? acc1 : acc0;
#pragma unroll
        for (int v = 0; v < 8; ++v) {
            const int co = mt * 16 + v + 8 * grp;
            if (co >= Cout) continue;
            float val = acc[v] + bias[co];
            if (relu) val = lrelu_f(val);
            if (out32) out32[out32A[s] + (long long)co * hw] = val;
            if (out16) out16[pixOutA[s] * CoutTotP + coutOff + co] = (_Float16)val;
        }
    }
}

// ---------------------------------------------------------------------------
// f32 NCHW channel-slice -> f16 NHWC mirror convert
// ---------------------------------------------------------------------------
__global__ __launch_bounds__(256) void cvt16_kernel(
    const float* __restrict__ src, int CtotS, int sOff,
    _Float16* __restrict__ dst, int CtotPD, int dOff,
    int B, int C, int H, int W)
{
    const int idx = blockIdx.x * blockDim.x + threadIdx.x;
    const int total = B * C * H * W;
    if (idx >= total) return;
    int t = idx;
    const int x = t % W; t /= W;
    const int y = t % H; t /= H;
    const int c = t % C; const int b = t / C;
    dst[((long long)(b * H + y) * W + x) * CtotPD + dOff + c] =
        (_Float16)src[(((long long)b * CtotS + sOff + c) * H + y) * W + x];
}

// ---------------------------------------------------------------------------
// ConvTranspose2d k=4 s=2 p=1, weights [Cin, Cout, 4, 4].  Cout == 2 always.
// ---------------------------------------------------------------------------
__global__ __launch_bounds__(256) void deconv4x4_kernel(
    const float* __restrict__ in, int B, int CinTot, int cinOff, int Cin,
    int Hin, int Win,
    const float* __restrict__ w, const float* __restrict__ bias,
    float* __restrict__ out, int CoutTot, int coutOff, int Cout,
    int Hout, int Wout)
{
    const int idx = blockIdx.x * blockDim.x + threadIdx.x;
    const int total = B * Cout * Hout * Wout;
    if (idx >= total) return;
    int t = idx;
    const int ox = t % Wout; t /= Wout;
    const int oy = t % Hout; t /= Hout;
    const int co = t % Cout; const int b = t / Cout;

    float acc = bias[co];
    const long long hwin = (long long)Hin * Win;
    const float* ib = in + ((long long)b * CinTot + cinOff) * hwin;
    for (int ky = 0; ky < 4; ++ky) {
        const int ys = oy + 1 - ky;
        if (ys & 1) continue;
        const int iy = ys >> 1;
        if ((unsigned)iy >= (unsigned)Hin) continue;
        for (int kx = 0; kx < 4; ++kx) {
            const int xs = ox + 1 - kx;
            if (xs & 1) continue;
            const int ix = xs >> 1;
            if ((unsigned)ix >= (unsigned)Win) continue;
            const float* ip = ib + (long long)iy * Win + ix;
            const float* wpt = w + (long long)co * 16 + ky * 4 + kx;
            for (int ci = 0; ci < Cin; ++ci)
                acc += ip[(long long)ci * hwin] * wpt[(long long)ci * Cout * 16];
        }
    }
    out[(((long long)b * CoutTot + coutOff + co) * Hout + oy) * Wout + ox] = acc;
}

// ---------------------------------------------------------------------------
// 81-displacement local cost volume, mean over C, LeakyReLU fused.
// ---------------------------------------------------------------------------
__global__ __launch_bounds__(256) void corr_kernel(
    const float* __restrict__ f1, const float* __restrict__ f2,
    int B, int C, int H, int W,
    float* __restrict__ dst, int CtotD, int dOff)
{
    const int idx = blockIdx.x * blockDim.x + threadIdx.x;
    const int total = B * 81 * H * W;
    if (idx >= total) return;
    int t = idx;
    const int x = t % W; t /= W;
    const int y = t % H; t /= H;
    const int d = t % 81; const int b = t / 81;
    const int dy = d / 9 - 4, dx = d % 9 - 4;
    const int y2 = y + dy, x2 = x + dx;

    float s = 0.f;
    if ((unsigned)y2 < (unsigned)H && (unsigned)x2 < (unsigned)W) {
        const long long hw = (long long)H * W;
        const float* a = f1 + (long long)b * C * hw + (long long)y * W + x;
        const float* p = f2 + (long long)b * C * hw + (long long)y2 * W + x2;
        for (int c = 0; c < C; ++c) s += a[c * hw] * p[c * hw];
    }
    s = lrelu_f(s / (float)C);
    dst[(((long long)b * CtotD + dOff + d) * H + y) * W + x] = s;
}

// ---------------------------------------------------------------------------
// Bicubic backwarp, border padding, align_corners=False.  With border padding
// the appended ones-channel samples to exactly 1 (cubic weights sum to 1), so
// the occlusion mask is identically 1 — dropped.
// ---------------------------------------------------------------------------
__device__ __forceinline__ void cubic_w(float t, float* w) {
    const float A = -0.75f;
    w[0] = ((A * (t + 1.f) - 5.f * A) * (t + 1.f) + 8.f * A) * (t + 1.f) - 4.f * A;
    w[1] = ((A + 2.f) * t - (A + 3.f)) * t * t + 1.f;
    const float u = 1.f - t;
    w[2] = ((A + 2.f) * u - (A + 3.f)) * u * u + 1.f;
    w[3] = 1.f - w[0] - w[1] - w[2];
}

__global__ __launch_bounds__(256) void backwarp_kernel(
    const float* __restrict__ img, int B, int C, int H, int W,
    const float* __restrict__ flow, int CtotF, int fOff, float dbl,
    float* __restrict__ dst)
{
    const int idx = blockIdx.x * blockDim.x + threadIdx.x;
    const int total = B * C * H * W;
    if (idx >= total) return;
    int t = idx;
    const int x = t % W; t /= W;
    const int y = t % H; t /= H;
    const int c = t % C; const int b = t / C;

    const float fx = flow[(((long long)b * CtotF + fOff + 0) * H + y) * W + x] * dbl;
    const float fy = flow[(((long long)b * CtotF + fOff + 1) * H + y) * W + x] * dbl;
    const float nx = -1.f + (2.f * x + 1.f) / W + fx * (2.f / (W - 1.f));
    const float ny = -1.f + (2.f * y + 1.f) / H + fy * (2.f / (H - 1.f));
    float ix = fminf(fmaxf((nx + 1.f) * W * 0.5f - 0.5f, 0.f), (float)(W - 1));
    float iy = fminf(fmaxf((ny + 1.f) * H * 0.5f - 0.5f, 0.f), (float)(H - 1));
    const float x0 = floorf(ix), y0 = floorf(iy);
    float wx[4], wy[4];
    cubic_w(ix - x0, wx);
    cubic_w(iy - y0, wy);
    int xs[4], ys[4];
#pragma unroll
    for (int k = 0; k < 4; ++k) {
        xs[k] = min(max((int)x0 - 1 + k, 0), W - 1);
        ys[k] = min(max((int)y0 - 1 + k, 0), H - 1);
    }
    const float* ip = img + ((long long)b * C + c) * H * W;
    float s = 0.f;
#pragma unroll
    for (int i = 0; i < 4; ++i) {
        float r = 0.f;
#pragma unroll
        for (int j = 0; j < 4; ++j) r += wx[j] * ip[(long long)ys[i] * W + xs[j]];
        s += wy[i] * r;
    }
    dst[idx] = s;
}

// ---------------------------------------------------------------------------
// channel-slice copy between channel-strided NCHW f32 buffers (concat fill)
// ---------------------------------------------------------------------------
__global__ __launch_bounds__(256) void copych_kernel(
    const float* __restrict__ src, int CtotS, int sOff,
    float* __restrict__ dst, int CtotD, int dOff,
    int B, int C, int H, int W)
{
    const int idx = blockIdx.x * blockDim.x + threadIdx.x;
    const int total = B * C * H * W;
    if (idx >= total) return;
    int t = idx;
    const int x = t % W; t /= W;
    const int y = t % H; t /= H;
    const int c = t % C; const int b = t / C;
    dst[(((long long)b * CtotD + dOff + c) * H + y) * W + x] =
        src[(((long long)b * CtotS + sOff + c) * H + y) * W + x];
}

__global__ __launch_bounds__(256) void add_kernel(
    const float* __restrict__ a, const float* __restrict__ b,
    float* __restrict__ o, int n)
{
    const int i = blockIdx.x * blockDim.x + threadIdx.x;
    if (i < n) o[i] = a[i] + b[i];
}

// ---------------------------------------------------------------------------
// half-pixel bilinear resize (jax.image.resize 'bilinear') with output scale
// ---------------------------------------------------------------------------
__global__ __launch_bounds__(256) void upsample_kernel(
    const float* __restrict__ in, int B, int C, int Hin, int Win,
    float* __restrict__ out, int Hout, int Wout, float scale)
{
    const int idx = blockIdx.x * blockDim.x + threadIdx.x;
    const int total = B * C * Hout * Wout;
    if (idx >= total) return;
    int t = idx;
    const int x = t % Wout; t /= Wout;
    const int y = t % Hout; t /= Hout;
    const int c = t % C; const int b = t / C;

    float sy = (y + 0.5f) * ((float)Hin / Hout) - 0.5f;
    float sx = (x + 0.5f) * ((float)Win / Wout) - 0.5f;
    sy = fminf(fmaxf(sy, 0.f), (float)(Hin - 1));
    sx = fminf(fmaxf(sx, 0.f), (float)(Win - 1));
    const int y0 = (int)floorf(sy), x0 = (int)floorf(sx);
    const int y1 = min(y0 + 1, Hin - 1), x1 = min(x0 + 1, Win - 1);
    const float ty = sy - y0, tx = sx - x0;
    const float* ip = in + ((long long)b * C + c) * Hin * Win;
    const float v =
        (1.f - ty) * ((1.f - tx) * ip[(long long)y0 * Win + x0] + tx * ip[(long long)y0 * Win + x1]) +
        ty         * ((1.f - tx) * ip[(long long)y1 * Win + x0] + tx * ip[(long long)y1 * Win + x1]);
    out[idx] = scale * v;
}

static inline int gsz(long long n) { return (int)((n + 255) / 256); }

extern "C" void kernel_launch(void* const* d_in, const int* in_sizes, int n_in,
                              void* d_out, int out_size, void* d_ws, size_t ws_size,
                              hipStream_t stream)
{
    (void)in_sizes; (void)n_in; (void)out_size; (void)ws_size;
    const float* frame1 = (const float*)d_in[0];
    const float* frame2 = (const float*)d_in[1];
    auto P = [&](int i) { return (const float*)d_in[i]; };
    // pytree flatten order (dict keys sorted):
    //   frames | decoders '2'..'6' | extractor list | refiner list
    // decoder base: cb0..cb5 | cw0..cw5 | uf_b uf_w up_b up_w  (L==6: no uf/up)
    auto decBase = [&](int L) { return 2 + (L - 2) * 16; };
    auto extIdx  = [&](int lev, int which) { return 78 + lev * 6 + which; }; // b0 b1 b2 w0 w1 w2
    auto refIdx  = [&](int j, int which) { return 114 + j * 2 + which; };   // b w

    // bump allocator over d_ws
    size_t used = 0;
    auto allocB = [&](size_t bytes) -> void* {
        void* p = (char*)d_ws + used;
        used += (bytes + 255) & ~(size_t)255;
        return p;
    };
    auto allocF = [&](size_t n) { return (float*)allocB(n * sizeof(float)); };
    auto allocH = [&](size_t n) { return (_Float16*)allocB(n * sizeof(_Float16)); };

    const int B = 2;
    const int chs[7] = {3, 16, 32, 64, 96, 128, 192};
    int Hs[7]; Hs[0] = 384;
    for (int i = 1; i <= 6; ++i) Hs[i] = Hs[i - 1] / 2;   // 192,96,48,24,12,6

    // conv launcher: packs weights (f16, tap-major padded K), then implicit GEMM
    auto conv = [&](const _Float16* in16, int CinTotP, int cinOff, int Cin,
                    int Hin, int Win, const float* w, const float* bias,
                    float* out32, int CoutTot,
                    _Float16* out16, int CoutTotP,
                    int coutOff, int Cout, int Hout, int Wout,
                    int stride, int pad, int dil, int relu) {
        const int Cpad = (Cin + 15) & ~15;
        const int Kp = 9 * Cpad;
        const int Mt = (Cout + 15) / 16;
        const int Kc = (Kp + 31) / 32;
        _Float16* wp = allocH((size_t)Mt * Kc * 32 * 16);
        packw_kernel<<<gsz((long long)Mt * Kc * 32), 256, 0, stream>>>(
            w, wp, Cout, Cin, Cpad, Mt, Kc);
        const int Ntot = B * Hout * Wout;
        const int Nt2 = (Ntot + 31) / 32;
        const int blocks = (int)(((long long)Mt * Nt2 + 7) / 8);
        conv3x3_wmma<<<blocks, 256, 0, stream>>>(in16, CinTotP, cinOff, Hin, Win,
            wp, bias, out32, CoutTot, out16, CoutTotP, coutOff, Cout, Hout, Wout,
            stride, pad, dil, relu, Mt, Ntot, Kc, Cpad);
    };

    // ---------------- feature pyramids ----------------
    float*     f1f[7];  float*     f2f[7];
    _Float16*  f1h[7];  _Float16*  f2h[7];
    for (int i = 1; i <= 6; ++i) {
        const size_t np = (size_t)B * Hs[i] * Hs[i];
        f1f[i] = allocF(np * chs[i]);  f1h[i] = allocH(np * (chs[i] + 16));
        f2f[i] = allocF(np * chs[i]);  f2h[i] = allocH(np * (chs[i] + 16));
    }
    _Float16* tAh = allocH(3000000);   // f16 NHWC ping-pong (extractor + refiner)
    _Float16* tBh = allocH(3000000);
    _Float16* frh = allocH((size_t)B * 384 * 384 * (3 + 16));  // frame mirror

    auto extract = [&](const float* frame, float** feats, _Float16** feath) {
        cvt16_kernel<<<gsz((long long)B * 3 * 384 * 384), 256, 0, stream>>>(
            frame, 3, 0, frh, 3 + 16, 0, B, 3, 384, 384);
        const _Float16* x = frh; int xs = 3 + 16;
        int cinPrev = 3, Hprev = 384;
        for (int i = 0; i < 6; ++i) {
            const int co = chs[i + 1], Ho = Hs[i + 1], sP = co + 16;
            conv(x, xs, 0, cinPrev, Hprev, Hprev, P(extIdx(i, 3)), P(extIdx(i, 0)),
                 nullptr, 0, tAh, sP, 0, co, Ho, Ho, 2, 1, 1, 1);
            conv(tAh, sP, 0, co, Ho, Ho, P(extIdx(i, 4)), P(extIdx(i, 1)),
                 nullptr, 0, tBh, sP, 0, co, Ho, Ho, 1, 1, 1, 1);
            conv(tBh, sP, 0, co, Ho, Ho, P(extIdx(i, 5)), P(extIdx(i, 2)),
                 feats[i + 1], co, feath[i + 1], sP, 0, co, Ho, Ho, 1, 1, 1, 1);
            x = feath[i + 1]; xs = sP; cinPrev = co; Hprev = Ho;
        }
    };
    extract(frame1, f1f, f1h);
    extract(frame2, f2f, f2h);

    // ---------------- coarse-to-fine decoders ----------------
    int curArr[7]; curArr[6] = 81; curArr[5] = 213; curArr[4] = 181; curArr[3] = 149; curArr[2] = 117;
    float dblArr[7]; dblArr[5] = 0.625f; dblArr[4] = 1.25f; dblArr[3] = 2.5f; dblArr[2] = 5.0f;
    float* warped = allocF((size_t)B * 32 * 96 * 96);   // max over levels (L=2)

    float* prevFlow = nullptr; float* prevFeat = nullptr;
    int prevCtot = 0, prevH = 0;
    _Float16* feath2 = nullptr; int feath2P = 0;   // level-2 mirror for refiner
    const int outs[6] = {128, 128, 96, 64, 32, 2};

    for (int L = 6; L >= 2; --L) {
        const int H = Hs[L], CL = chs[L];
        const int cur = curArr[L], Ctot = cur + 448, CtotP = Ctot + 16;
        const int base = decBase(L);
        float*    feat   = allocF((size_t)B * Ctot * H * H);
        _Float16* feat16 = allocH((size_t)B * H * H * CtotP);
        float*    flow   = allocF((size_t)B * 2 * H * H);

        if (L == 6) {
            corr_kernel<<<gsz((long long)B * 81 * H * H), 256, 0, stream>>>(
                f1f[6], f2f[6], B, CL, H, H, feat, Ctot, 448);
        } else {
            const int firstOff = 529;             // vol(81) @448, first(CL) @529
            const int flowOff  = 529 + CL;        // flow_up(2)
            const int featOff  = flowOff + 2;     // feat_up(2)
            deconv4x4_kernel<<<gsz((long long)B * 2 * H * H), 256, 0, stream>>>(
                prevFlow, B, 2, 0, 2, prevH, prevH, P(base + 13), P(base + 12),
                feat, Ctot, flowOff, 2, H, H);
            deconv4x4_kernel<<<gsz((long long)B * 2 * H * H), 256, 0, stream>>>(
                prevFeat, B, prevCtot, 0, prevCtot, prevH, prevH, P(base + 15), P(base + 14),
                feat, Ctot, featOff, 2, H, H);
            backwarp_kernel<<<gsz((long long)B * CL * H * H), 256, 0, stream>>>(
                f2f[L], B, CL, H, H, feat, Ctot, flowOff, dblArr[L], warped);
            corr_kernel<<<gsz((long long)B * 81 * H * H), 256, 0, stream>>>(
                f1f[L], warped, B, CL, H, H, feat, Ctot, 448);
            copych_kernel<<<gsz((long long)B * CL * H * H), 256, 0, stream>>>(
                f1f[L], CL, 0, feat, Ctot, firstOff, B, CL, H, H);
        }
        // f16 NHWC mirror of the scalar-produced base region [448, 448+cur)
        cvt16_kernel<<<gsz((long long)B * cur * H * H), 256, 0, stream>>>(
            feat, Ctot, 448, feat16, CtotP, 448, B, cur, H, H);

        // dense conv stack, prepended outputs at descending channel offsets
        int roff = 448, cin = cur;
        for (int k = 0; k < 5; ++k) {
            const int co = outs[k];
            conv(feat16, CtotP, roff, cin, H, H, P(base + 6 + k), P(base + k),
                 feat, Ctot, feat16, CtotP, roff - co, co, H, H, 1, 1, 1, 1);
            roff -= co; cin += co;
        }
        // flow head (no activation)
        conv(feat16, CtotP, 0, Ctot, H, H, P(base + 6 + 5), P(base + 5),
             flow, 2, nullptr, 0, 0, 2, H, H, 1, 1, 1, 0);

        prevFlow = flow; prevFeat = feat; prevCtot = Ctot; prevH = H;
        feath2 = feat16; feath2P = CtotP;
    }

    // ---------------- context refiner (dilated convs) ----------------
    const int rdil[7] = {1, 2, 4, 8, 16, 1, 1};
    const int rco[7]  = {128, 128, 128, 96, 64, 32, 2};
    float* refOut = allocF((size_t)B * 2 * 96 * 96);
    const _Float16* rx = feath2; int rxP = feath2P; int rCin = prevCtot;
    for (int j = 0; j < 7; ++j) {
        _Float16* o16 = (j == 6) ? nullptr : ((j & 1) ? tBh : tAh);
        const int oP  = rco[j] + 16;
        float*    o32 = (j == 6) ? refOut : nullptr;
        conv(rx, rxP, 0, rCin, 96, 96, P(refIdx(j, 1)), P(refIdx(j, 0)),
             o32, rco[j], o16, oP, 0, rco[j], 96, 96, 1, rdil[j], rdil[j], (j < 6) ? 1 : 0);
        rx = o16; rxP = oP; rCin = rco[j];
    }

    // flow + refinement, then x4 half-pixel bilinear upsample scaled by 20
    float* flowFinal = allocF((size_t)B * 2 * 96 * 96);
    add_kernel<<<gsz((long long)B * 2 * 96 * 96), 256, 0, stream>>>(
        prevFlow, refOut, flowFinal, B * 2 * 96 * 96);
    upsample_kernel<<<gsz((long long)B * 2 * 384 * 384), 256, 0, stream>>>(
        flowFinal, B, 2, 96, 96, (float*)d_out, 384, 384, 20.0f);
}
